// SelfBlock_45603962749380
// MI455X (gfx1250) — matrix operand order repaired and checked by reading, not compiled, not run on hardware
//
#include <hip/hip_runtime.h>
#include <cstdint>

// ---------------- problem constants (from reference) ----------------
static constexpr int Bq = 2, Nq = 2048, Eq = 1024, Hq = 16, Dq = 64;
static constexpr int Mrows = Bq * Nq;   // 4096 token rows
static constexpr int E3 = 3 * Eq;       // 3072
static constexpr int E2 = 2 * Eq;       // 2048

// ---------------- vector types for WMMA ----------------
typedef __attribute__((ext_vector_type(16))) __bf16 v16bf;
typedef __attribute__((ext_vector_type(8)))  __bf16 v8bf;
typedef __attribute__((ext_vector_type(8)))  float  v8f;

#define WMMA_BF16(a, b, c) \
  __builtin_amdgcn_wmma_f32_16x16x32_bf16(false, (a), false, (b), (short)0, (c), false, false)

// float -> bf16 round-to-nearest-even (bit-exact, no reliance on native cvt)
__device__ __forceinline__ __bf16 f2bf(float f) {
  uint32_t u = __float_as_uint(f);
  uint32_t r = u + 0x7FFFu + ((u >> 16) & 1u);
  return __builtin_bit_cast(__bf16, (uint16_t)(r >> 16));
}
__device__ __forceinline__ float bf2f(__bf16 h) {
  uint32_t u = ((uint32_t)__builtin_bit_cast(uint16_t, h)) << 16;
  return __uint_as_float(u);
}

// Build a 16-element bf16 fragment from two contiguous 16-byte chunks.
__device__ __forceinline__ v16bf mk16(const __bf16* p, const __bf16* q) {
  v8bf a = *(const v8bf*)p;
  v8bf b = *(const v8bf*)q;
  v16bf r;
#pragma unroll
  for (int i = 0; i < 8; ++i) { r[i] = a[i]; r[i + 8] = b[i]; }
  return r;
}

// ================= split f32 -> (bf16 hi, bf16 lo) =================
__global__ __launch_bounds__(256) void split2d_kernel(
    const float* __restrict__ src, int sld,
    __bf16* __restrict__ dh, __bf16* __restrict__ dl, int dld,
    int rows, int cols) {
  int idx = blockIdx.x * 256 + threadIdx.x;
  if (idx >= rows * cols) return;
  int r = idx / cols, c = idx - r * cols;
  float v = src[(size_t)r * sld + c];
  __bf16 hi = f2bf(v);
  dh[(size_t)r * dld + c] = hi;
  dl[(size_t)r * dld + c] = f2bf(v - bf2f(hi));
}

// ================= bf16x3 WMMA GEMM =================
// C[M,N] = A[M,K] * W[N,K]^T + bias, A/W given as hi+lo bf16 splits.
// Wave tile: 16 rows x 64 cols (4 accumulators). Block: 8 waves -> 128x64.
__global__ __launch_bounds__(256) void gemm_bf16x3_kernel(
    const __bf16* __restrict__ Ah, const __bf16* __restrict__ Al, int lda,
    const __bf16* __restrict__ Wh, const __bf16* __restrict__ Wl, int ldw,
    const float* __restrict__ bias,
    float* __restrict__ Cf, int ldc,
    __bf16* __restrict__ Oh, __bf16* __restrict__ Ol, int ldo,
    const float* __restrict__ resid, int ldr,
    int K) {
  const int lane = threadIdx.x & 31;
  const int wid  = threadIdx.x >> 5;
  const int lp   = lane & 15;
  const int half = lane >> 4;
  const int mBase = blockIdx.y * 128 + wid * 16;
  const int nBase = blockIdx.x * 64;

  const v8f z8 = {0.f, 0.f, 0.f, 0.f, 0.f, 0.f, 0.f, 0.f};
  v8f acc[4] = {z8, z8, z8, z8};

  const __bf16* arow_h = Ah + (size_t)(mBase + lp) * lda + 8 * half;
  const __bf16* arow_l = Al + (size_t)(mBase + lp) * lda + 8 * half;

  for (int kk = 0; kk < K; kk += 32) {
    v16bf a_h = mk16(arow_h + kk, arow_h + kk + 16);
    v16bf a_l = mk16(arow_l + kk, arow_l + kk + 16);
#pragma unroll
    for (int t = 0; t < 4; ++t) {
      const __bf16* wp_h = Wh + (size_t)(nBase + t * 16 + lp) * ldw + kk + 8 * half;
      const __bf16* wp_l = Wl + (size_t)(nBase + t * 16 + lp) * ldw + kk + 8 * half;
      v16bf b_h = mk16(wp_h, wp_h + 16);
      v16bf b_l = mk16(wp_l, wp_l + 16);
      acc[t] = WMMA_BF16(a_h, b_h, acc[t]);
      acc[t] = WMMA_BF16(a_h, b_l, acc[t]);
      acc[t] = WMMA_BF16(a_l, b_h, acc[t]);
    }
  }

  const int rowOff = mBase + 8 * half;
#pragma unroll
  for (int t = 0; t < 4; ++t) {
    int col = nBase + t * 16 + lp;
    float bv = bias ? bias[col] : 0.f;
#pragma unroll
    for (int r = 0; r < 8; ++r) {
      int row = rowOff + r;
      float v = acc[t][r] + bv;
      if (resid) v += resid[(size_t)row * ldr + col];
      if (Cf) Cf[(size_t)row * ldc + col] = v;
      if (Oh) {
        __bf16 hi = f2bf(v);
        Oh[(size_t)row * ldo + col] = hi;
        Ol[(size_t)row * ldo + col] = f2bf(v - bf2f(hi));
      }
    }
  }
}

// ================= RoPE + head split + V transpose =================
// qkv f32 [B,N,3E] with f = h*D*3 + d*3 + comp  (per reference reshape)
// -> q,k (roped, bf16 hi/lo, [B,H,N,D]) and V^T (bf16 hi/lo, [B,H,D,N])
__global__ __launch_bounds__(256) void rope_split_kernel(
    const float* __restrict__ qkv, const float* __restrict__ enc,
    __bf16* __restrict__ qh, __bf16* __restrict__ ql,
    __bf16* __restrict__ kh, __bf16* __restrict__ kl,
    __bf16* __restrict__ vth, __bf16* __restrict__ vtl) {
  int idx = blockIdx.x * 256 + threadIdx.x;
  if (idx >= Bq * Hq * Nq * Dq) return;
  int d  = idx & (Dq - 1);
  int n  = (idx >> 6) & (Nq - 1);
  int hh = (idx >> 17) & (Hq - 1);
  int b  = idx >> 21;

  size_t rowoff = (size_t)(b * Nq + n) * E3;
  int fb = hh * Dq * 3 + d * 3;
  float q = qkv[rowoff + fb + 0];
  float k = qkv[rowoff + fb + 1];
  float v = qkv[rowoff + fb + 2];
  int fp = hh * Dq * 3 + (d ^ 1) * 3;
  float qp = qkv[rowoff + fp + 0];
  float kp = qkv[rowoff + fp + 1];
  float rq = (d & 1) ? qp : -qp;   // rotate_half: even d -> -t[d+1], odd d -> t[d-1]
  float rk = (d & 1) ? kp : -kp;

  size_t eoff = (size_t)(b * Nq + n) * Dq + d;
  float e0 = enc[eoff];
  float e1 = enc[(size_t)Bq * Nq * Dq + eoff];
  float qr = q * e0 + rq * e1;
  float kr = k * e0 + rk * e1;

  size_t qi = (size_t)((b * Hq + hh) * Nq + n) * Dq + d;
  __bf16 t;
  t = f2bf(qr); qh[qi] = t; ql[qi] = f2bf(qr - bf2f(t));
  t = f2bf(kr); kh[qi] = t; kl[qi] = f2bf(kr - bf2f(t));
  size_t vi = (size_t)((b * Hq + hh) * Dq + d) * Nq + n;
  t = f2bf(v);  vth[vi] = t; vtl[vi] = f2bf(v - bf2f(t));
}

// ================= flash attention (WMMA, online softmax) =================
// One wave per (b, h, 16-row query tile). 8 waves / block.
__global__ __launch_bounds__(256) void attn_kernel(
    const __bf16* __restrict__ qh, const __bf16* __restrict__ ql,
    const __bf16* __restrict__ kh, const __bf16* __restrict__ kl,
    const __bf16* __restrict__ vth, const __bf16* __restrict__ vtl,
    __bf16* __restrict__ ctxh, __bf16* __restrict__ ctxl) {
  __shared__ __bf16 sPh[8][16 * 32];
  __shared__ __bf16 sPl[8][16 * 32];

  const int lane = threadIdx.x & 31;
  const int wid  = threadIdx.x >> 5;
  const int lp   = lane & 15;
  const int half = lane >> 4;

  int gid = blockIdx.x * 8 + wid;          // 4096 tiles total
  int it = gid & 127;
  int h  = (gid >> 7) & 15;
  int b  = gid >> 11;
  int i0 = it * 16;

  size_t hoff = (size_t)(b * Hq + h) * Nq * Dq;  // same for q,k and v^T
  const __bf16* Qh_ = qh + hoff;  const __bf16* Ql_ = ql + hoff;
  const __bf16* Kh_ = kh + hoff;  const __bf16* Kl_ = kl + hoff;
  const __bf16* Vth_ = vth + hoff; const __bf16* Vtl_ = vtl + hoff;

  // preload Q fragments (rows i0..i0+15, K = head dim 64 in two 32-steps)
  const __bf16* qrow_h = Qh_ + (size_t)(i0 + lp) * Dq + 8 * half;
  const __bf16* qrow_l = Ql_ + (size_t)(i0 + lp) * Dq + 8 * half;
  v16bf qfh[2], qfl[2];
#pragma unroll
  for (int e = 0; e < 2; ++e) {
    qfh[e] = mk16(qrow_h + e * 32, qrow_h + e * 32 + 16);
    qfl[e] = mk16(qrow_l + e * 32, qrow_l + e * 32 + 16);
  }

  const v8f z8 = {0.f, 0.f, 0.f, 0.f, 0.f, 0.f, 0.f, 0.f};
  v8f O[4] = {z8, z8, z8, z8};
  float mrow[8], lrow[8];
#pragma unroll
  for (int r = 0; r < 8; ++r) { mrow[r] = -3.0e38f; lrow[r] = 0.f; }

  const float sc = 0.125f * 1.4426950408889634f;  // D^-0.5 * log2(e)

  for (int j0 = 0; j0 < Nq; j0 += 32) {
    asm volatile("s_wait_dscnt 0" ::: "memory");  // prior-iter LDS reads done

    v8f S0 = z8, S1 = z8;
#pragma unroll
    for (int e = 0; e < 2; ++e) {
      int kk = e * 32;
      {
        const __bf16* ph = Kh_ + (size_t)(j0 + lp) * Dq + kk + 8 * half;
        const __bf16* pl = Kl_ + (size_t)(j0 + lp) * Dq + kk + 8 * half;
        v16bf bh = mk16(ph, ph + 16), bl = mk16(pl, pl + 16);
        S0 = WMMA_BF16(qfh[e], bh, S0);
        S0 = WMMA_BF16(qfh[e], bl, S0);
        S0 = WMMA_BF16(qfl[e], bh, S0);
      }
      {
        const __bf16* ph = Kh_ + (size_t)(j0 + 16 + lp) * Dq + kk + 8 * half;
        const __bf16* pl = Kl_ + (size_t)(j0 + 16 + lp) * Dq + kk + 8 * half;
        v16bf bh = mk16(ph, ph + 16), bl = mk16(pl, pl + 16);
        S1 = WMMA_BF16(qfh[e], bh, S1);
        S1 = WMMA_BF16(qfh[e], bl, S1);
        S1 = WMMA_BF16(qfl[e], bh, S1);
      }
    }

    // online softmax in C-fragment layout: lane holds rows r+8*half, col lp
#pragma unroll
    for (int r = 0; r < 8; ++r) {
      float t0 = S0[r] * sc, t1 = S1[r] * sc;
      float mx = fmaxf(t0, t1);
#pragma unroll
      for (int msk = 1; msk <= 8; msk <<= 1) mx = fmaxf(mx, __shfl_xor(mx, msk, 32));
      float mnew  = fmaxf(mrow[r], mx);
      float alpha = exp2f(mrow[r] - mnew);
      float p0 = exp2f(t0 - mnew), p1 = exp2f(t1 - mnew);
      float rs = p0 + p1;
#pragma unroll
      for (int msk = 1; msk <= 8; msk <<= 1) rs += __shfl_xor(rs, msk, 32);
      lrow[r] = lrow[r] * alpha + rs;
      mrow[r] = mnew;
#pragma unroll
      for (int dt = 0; dt < 4; ++dt) O[dt][r] *= alpha;

      int row = r + 8 * half;
      __bf16 h0 = f2bf(p0);
      sPh[wid][row * 32 + lp] = h0;
      sPl[wid][row * 32 + lp] = f2bf(p0 - bf2f(h0));
      __bf16 h1 = f2bf(p1);
      sPh[wid][row * 32 + 16 + lp] = h1;
      sPl[wid][row * 32 + 16 + lp] = f2bf(p1 - bf2f(h1));
    }

    asm volatile("s_wait_dscnt 0" ::: "memory");  // LDS stores visible (wave lockstep)
    v16bf pfh = mk16(&sPh[wid][lp * 32 + 8 * half], &sPh[wid][lp * 32 + 8 * half + 16]);
    v16bf pfl = mk16(&sPl[wid][lp * 32 + 8 * half], &sPl[wid][lp * 32 + 8 * half + 16]);

#pragma unroll
    for (int dt = 0; dt < 4; ++dt) {
      const __bf16* ph = Vth_ + (size_t)(dt * 16 + lp) * Nq + j0 + 8 * half;
      const __bf16* pl = Vtl_ + (size_t)(dt * 16 + lp) * Nq + j0 + 8 * half;
      v16bf vbh = mk16(ph, ph + 16), vbl = mk16(pl, pl + 16);
      O[dt] = WMMA_BF16(pfh, vbh, O[dt]);
      O[dt] = WMMA_BF16(pfh, vbl, O[dt]);
      O[dt] = WMMA_BF16(pfl, vbh, O[dt]);
    }
  }

  // finalize: ctx[b, tok, h*D + d] as bf16 hi/lo for the out-proj GEMM
#pragma unroll
  for (int r = 0; r < 8; ++r) {
    float inv = 1.0f / lrow[r];
    int tok = i0 + r + 8 * half;
#pragma unroll
    for (int dt = 0; dt < 4; ++dt) {
      float v = O[dt][r] * inv;
      int e = h * Dq + dt * 16 + lp;
      size_t idx = (size_t)(b * Nq + tok) * Eq + e;
      __bf16 hi = f2bf(v);
      ctxh[idx] = hi;
      ctxl[idx] = f2bf(v - bf2f(hi));
    }
  }
}

// ================= LayerNorm + exact GELU -> bf16 split =================
__global__ __launch_bounds__(256) void ln_gelu_kernel(
    const float* __restrict__ X, const float* __restrict__ g,
    const float* __restrict__ bta,
    __bf16* __restrict__ dh, __bf16* __restrict__ dl) {
  __shared__ float red[256];
  int row = blockIdx.x, t = threadIdx.x;
  const float* xr = X + (size_t)row * E2;
  float v[8];
  float s = 0.f;
#pragma unroll
  for (int i = 0; i < 8; ++i) { v[i] = xr[t + i * 256]; s += v[i]; }
  red[t] = s; __syncthreads();
  for (int o = 128; o > 0; o >>= 1) { if (t < o) red[t] += red[t + o]; __syncthreads(); }
  float mu = red[0] * (1.0f / E2);
  __syncthreads();
  float s2 = 0.f;
#pragma unroll
  for (int i = 0; i < 8; ++i) { float d = v[i] - mu; s2 += d * d; }
  red[t] = s2; __syncthreads();
  for (int o = 128; o > 0; o >>= 1) { if (t < o) red[t] += red[t + o]; __syncthreads(); }
  float rstd = rsqrtf(red[0] * (1.0f / E2) + 1e-5f);
#pragma unroll
  for (int i = 0; i < 8; ++i) {
    int c = t + i * 256;
    float y = (v[i] - mu) * rstd * g[c] + bta[c];
    float ge = 0.5f * y * (1.f + erff(y * 0.70710678118654752f));
    __bf16 hi = f2bf(ge);
    dh[(size_t)row * E2 + c] = hi;
    dl[(size_t)row * E2 + c] = f2bf(ge - bf2f(hi));
  }
}

// =========================== host orchestration ===========================
extern "C" void kernel_launch(void* const* d_in, const int* in_sizes, int n_in,
                              void* d_out, int out_size, void* d_ws, size_t ws_size,
                              hipStream_t stream) {
  (void)in_sizes; (void)n_in; (void)out_size; (void)ws_size;
  const float* x      = (const float*)d_in[0];
  const float* enc    = (const float*)d_in[1];
  const float* Wqkv_w = (const float*)d_in[2];
  const float* Wqkv_b = (const float*)d_in[3];
  const float* out_w  = (const float*)d_in[4];
  const float* out_b  = (const float*)d_in[5];
  const float* ffn1_w = (const float*)d_in[6];
  const float* ffn1_b = (const float*)d_in[7];
  const float* ln_g   = (const float*)d_in[8];
  const float* ln_b   = (const float*)d_in[9];
  const float* ffn2_w = (const float*)d_in[10];
  const float* ffn2_b = (const float*)d_in[11];
  float* out = (float*)d_out;

  // workspace carve-up (≈193 MB, with region reuse)
  size_t off = 0;
  auto nextp = [&](size_t bytes) -> char* {
    char* p = (char*)d_ws + off;
    off += (bytes + 255) & ~(size_t)255;
    return p;
  };
  const size_t BHND = (size_t)Bq * Hq * Nq * Dq;  // 4,194,304

  __bf16* h2h = (__bf16*)nextp((size_t)Mrows * E2 * 2);  // concat [x | message] hi
  __bf16* h2l = (__bf16*)nextp((size_t)Mrows * E2 * 2);  //                     lo
  __bf16* wqh = (__bf16*)nextp((size_t)E3 * Eq * 2);
  __bf16* wql = (__bf16*)nextp((size_t)E3 * Eq * 2);
  __bf16* owh = (__bf16*)nextp((size_t)Eq * Eq * 2);
  __bf16* owl = (__bf16*)nextp((size_t)Eq * Eq * 2);
  __bf16* f1h = (__bf16*)nextp((size_t)E2 * E2 * 2);
  __bf16* f1l = (__bf16*)nextp((size_t)E2 * E2 * 2);
  __bf16* f2h = (__bf16*)nextp((size_t)Eq * E2 * 2);
  __bf16* f2l = (__bf16*)nextp((size_t)Eq * E2 * 2);
  char*  regC = nextp((size_t)Mrows * E3 * 4);           // qkv f32, reused for ffn1 out
  float* qkvf = (float*)regC;
  float* hbuf = (float*)regC;                            // [Mrows, 2E] fits inside regC
  char*  regD = (char*)d_ws + off;                       // q/k/vt, reused for gelu out
  __bf16* q_h = (__bf16*)nextp(BHND * 2);
  __bf16* q_l = (__bf16*)nextp(BHND * 2);
  __bf16* k_h = (__bf16*)nextp(BHND * 2);
  __bf16* k_l = (__bf16*)nextp(BHND * 2);
  __bf16* vt_h = (__bf16*)nextp(BHND * 2);
  __bf16* vt_l = (__bf16*)nextp(BHND * 2);
  __bf16* g_h = (__bf16*)regD;                           // reuse after attention
  __bf16* g_l = (__bf16*)(regD + (size_t)Mrows * E2 * 2);
  __bf16* ctxh = (__bf16*)nextp((size_t)Mrows * Eq * 2);
  __bf16* ctxl = (__bf16*)nextp((size_t)Mrows * Eq * 2);

  auto blocks = [](size_t n) { return (unsigned)((n + 255) / 256); };

  // 1) splits: x into concat-buffer cols [0,E); weights
  split2d_kernel<<<blocks((size_t)Mrows * Eq), 256, 0, stream>>>(x, Eq, h2h, h2l, E2, Mrows, Eq);
  split2d_kernel<<<blocks((size_t)E3 * Eq), 256, 0, stream>>>(Wqkv_w, Eq, wqh, wql, Eq, E3, Eq);
  split2d_kernel<<<blocks((size_t)Eq * Eq), 256, 0, stream>>>(out_w, Eq, owh, owl, Eq, Eq, Eq);
  split2d_kernel<<<blocks((size_t)E2 * E2), 256, 0, stream>>>(ffn1_w, E2, f1h, f1l, E2, E2, E2);
  split2d_kernel<<<blocks((size_t)Eq * E2), 256, 0, stream>>>(ffn2_w, E2, f2h, f2l, E2, Eq, E2);

  // 2) qkv = x @ Wqkv^T + b   -> f32 [4096,3072]
  gemm_bf16x3_kernel<<<dim3(E3 / 64, Mrows / 128), 256, 0, stream>>>(
      h2h, h2l, E2, wqh, wql, Eq, Wqkv_b,
      qkvf, E3, nullptr, nullptr, 0, nullptr, 0, Eq);

  // 3) RoPE + head split + V transpose
  rope_split_kernel<<<blocks(BHND), 256, 0, stream>>>(qkvf, enc, q_h, q_l, k_h, k_l, vt_h, vt_l);

  // 4) flash attention -> ctx bf16 hi/lo [4096,1024]
  attn_kernel<<<(Bq * Hq * (Nq / 16)) / 8, 256, 0, stream>>>(
      q_h, q_l, k_h, k_l, vt_h, vt_l, ctxh, ctxl);

  // 5) message = ctx @ out_w^T + out_b -> bf16 split into concat cols [E,2E)
  gemm_bf16x3_kernel<<<dim3(Eq / 64, Mrows / 128), 256, 0, stream>>>(
      ctxh, ctxl, Eq, owh, owl, Eq, out_b,
      nullptr, 0, h2h + Eq, h2l + Eq, E2, nullptr, 0, Eq);

  // 6) h = concat @ ffn1_w^T + b -> f32 [4096,2048]
  gemm_bf16x3_kernel<<<dim3(E2 / 64, Mrows / 128), 256, 0, stream>>>(
      h2h, h2l, E2, f1h, f1l, E2, ffn1_b,
      hbuf, E2, nullptr, nullptr, 0, nullptr, 0, E2);

  // 7) layernorm + exact gelu -> bf16 split
  ln_gelu_kernel<<<Mrows, 256, 0, stream>>>(hbuf, ln_g, ln_b, g_h, g_l);

  // 8) out = x + g @ ffn2_w^T + b -> f32 d_out
  gemm_bf16x3_kernel<<<dim3(Eq / 64, Mrows / 128), 256, 0, stream>>>(
      g_h, g_l, E2, f2h, f2l, E2, ffn2_b,
      out, Eq, nullptr, nullptr, 0, x, Eq, E2);
}
